// StereographicAttentionLayer_89232240541713
// MI455X (gfx1250) — compile-verified, matrix operand order checked by reading, not run on hardware
//
#include <hip/hip_runtime.h>

#define NN   16384
#define DIMC 1024
#define HC   16
#define DHC  64
#define EPSF 1e-15f
#define NSLICE 32

typedef __attribute__((ext_vector_type(16))) _Float16 v16h;
typedef __attribute__((ext_vector_type(8)))  float    v8f;
typedef __attribute__((ext_vector_type(4)))  float    v4f;

union AF { v16h v; v4f f4[2]; };

__device__ inline float eluf(float x) { return x > 0.0f ? x : expm1f(x); }

__device__ inline float artk_sk(float x, float sk) {
  float z = fminf(fmaxf(x * sk, -1.0f + 1e-7f), 1.0f - 1e-7f);
  return atanhf(z) / sk;
}

// scalar chain for: project(expmap0(mu,k)) -> mobius_add(.,sb) -> project
// returns p,q such that result = fp2*(p*mu + q*sb); vv = ||result||^2
__device__ inline void stereo_post(float S_mm, float S_ms, float kv, float y2,
                                   float& pO, float& qO, float& f2O, float& vvO) {
  float sk   = sqrtf(fmaxf(fabsf(kv), EPSF));
  float un   = fmaxf(sqrtf(fmaxf(S_mm, 0.0f)), EPSF);
  float tk   = tanhf(un * sk) / sk;
  float fexp = tk / un;
  float rn   = fabsf(tk);
  float maxn = (1.0f - 1e-5f) / sk;
  float n1   = fmaxf(rn, EPSF);
  float fp1  = (n1 > maxn) ? maxn / n1 : 1.0f;
  float a    = fexp * fp1;
  float x2   = a * a * S_mm;
  float xy   = a * S_ms;
  float alpha = 1.0f - 2.0f * kv * xy - kv * y2;
  float beta  = 1.0f + kv * x2;
  float dm    = fmaxf(1.0f - 2.0f * kv * xy + kv * kv * x2 * y2, EPSF);
  float p = alpha * a / dm;
  float q = beta / dm;
  float vn2 = p * p * S_mm + 2.0f * p * q * S_ms + q * q * y2;
  float r2  = sqrtf(fmaxf(vn2, 0.0f));
  float n2  = fmaxf(r2, EPSF);
  float fp2 = (n2 > maxn) ? maxn / n2 : 1.0f;
  pO = p; qO = q; f2O = fp2; vvO = fp2 * fp2 * vn2;
}

// two simultaneous sums over a 64-thread (2-wave) block, result broadcast
__device__ inline float2 blockReduce64x2(float a, float b) {
  __shared__ float buf[4];
  #pragma unroll
  for (int off = 16; off > 0; off >>= 1) {
    a += __shfl_down(a, off, 32);
    b += __shfl_down(b, off, 32);
  }
  int t = threadIdx.x;
  if ((t & 31) == 0) { buf[(t >> 5) * 2] = a; buf[(t >> 5) * 2 + 1] = b; }
  __syncthreads();
  float2 r; r.x = buf[0] + buf[2]; r.y = buf[1] + buf[3];
  __syncthreads();
  return r;
}

// ---- WMMA fragment helpers --------------------------------------------------
// A (16x32 f16) and B (32x16 f16, sourced from row-major [out][K]) fragments
// are each two contiguous 16B chunks per lane: chunk base kh = (lane>>4)*8,
// second chunk at kh+16.
__device__ inline void loadA(AF& a, const _Float16* aRow, int k) {
  a.f4[0] = *(const v4f*)(aRow + k);
  a.f4[1] = *(const v4f*)(aRow + k + 16);
}
__device__ inline void loadB4(AF* b, const _Float16* wBase, long long ldw, int k) {
  #pragma unroll
  for (int j = 0; j < 4; ++j) {
    const _Float16* wRow = wBase + (long long)(j * 16) * ldw + k;
    b[j].f4[0] = *(const v4f*)(wRow);
    b[j].f4[1] = *(const v4f*)(wRow + 16);
  }
}
__device__ inline void mma4(v8f* acc, const AF& a, const AF* b) {
  #pragma unroll
  for (int j = 0; j < 4; ++j)
    acc[j] = __builtin_amdgcn_wmma_f32_16x16x32_f16(
        false, a.v, false, b[j].v, (short)0, acc[j], false, false);
}

__global__ __launch_bounds__(256) void cvt_f32_f16(const float* __restrict__ in,
                                                   _Float16* __restrict__ out, int n) {
  int i = blockIdx.x * 256 + threadIdx.x;
  if (i < n) out[i] = (_Float16)in[i];
}

// sb = project(expmap0(bias_head, k)); also ||sb||^2 per head
__global__ __launch_bounds__(64) void sb_kernel(
    const float* __restrict__ bv, const float* __restrict__ cv,
    const float* __restrict__ bff, const float* __restrict__ cff,
    float* __restrict__ sbv, float* __restrict__ sbv2,
    float* __restrict__ sbff, float* __restrict__ sbff2) {
  int h = blockIdx.x, which = blockIdx.y, d = threadIdx.x;
  const float* b = which ? bff : bv;
  const float* c = which ? cff : cv;
  float* sb  = which ? sbff : sbv;
  float* sb2 = which ? sbff2 : sbv2;
  float x = b[h * DHC + d];
  float2 rr = blockReduce64x2(x * x, 0.0f);
  float S = rr.x;
  float k = c[h];
  float sk = sqrtf(fmaxf(fabsf(k), EPSF));
  float un = fmaxf(sqrtf(fmaxf(S, 0.0f)), EPSF);
  float tk = tanhf(un * sk) / sk;
  float fexp = tk / un;
  float rn = fabsf(tk);
  float maxn = (1.0f - 1e-5f) / sk;
  float n1 = fmaxf(rn, EPSF);
  float fp = (n1 > maxn) ? maxn / n1 : 1.0f;
  float scale = fexp * fp;
  sb[h * DHC + d] = scale * x;
  if (d == 0) sb2[h] = scale * scale * S;
}

// per (n,h): Uh = f16(logmap0(X,c_v)); also Xh = f16(X)
__global__ __launch_bounds__(64) void logmap_x_kernel(
    const float* __restrict__ X, const float* __restrict__ c_v,
    _Float16* __restrict__ Xh, _Float16* __restrict__ Uh) {
  int n = blockIdx.x, h = blockIdx.y, d = threadIdx.x;
  long long idx = (long long)n * DIMC + h * DHC + d;
  float x = X[idx];
  Xh[idx] = (_Float16)x;
  float2 rr = blockReduce64x2(x * x, 0.0f);
  float yn = fmaxf(sqrtf(fmaxf(rr.x, 0.0f)), EPSF);
  float k = c_v[h];
  float sk = sqrtf(fmaxf(fabsf(k), EPSF));
  float u = artk_sk(yn, sk) / yn * x;
  Uh[idx] = (_Float16)u;
}

// C[m][n0+j*16+c] (+bias) = A(MxK,row-major) x W(NoutxK,row-major)^T
// one wave: 16(M) x 64(N) tile; block = 4 waves over M; batched via blockIdx.z.
// Double-buffered K pipeline (K must be a multiple of 64): loads for the next
// 32-wide K step are in flight while the current step's WMMAs execute, so the
// compiler can use partial s_wait_loadcnt instead of draining VMEM per WMMA.
__global__ __launch_bounds__(128) void gemm_wmma_kernel(
    const _Float16* __restrict__ A, int lda, long long sA,
    const _Float16* __restrict__ W, int ldw, long long sW,
    float* __restrict__ C, int ldc, long long sC,
    const float* __restrict__ bias, int M, int K) {
  int wave = threadIdx.x >> 5;
  int lane = threadIdx.x & 31;
  long long bz = blockIdx.z;
  A += bz * sA; W += bz * sW; C += bz * sC;
  int m0 = (blockIdx.x * 4 + wave) * 16;
  if (m0 >= M) return;
  int n0 = blockIdx.y * 64;
  int lr = lane & 15;
  int kh = (lane >> 4) << 3;   // K chunk base: 0 or 8 (chunks at kh, kh+16)
  const _Float16* aRow  = A + (long long)(m0 + lr) * lda + kh;
  const _Float16* wBase = W + (long long)(n0 + lr) * ldw + kh;
  v8f z = {};
  v8f acc[4] = {z, z, z, z};
  AF a0, a1;
  AF b0[4], b1[4];
  loadA(a0, aRow, 0);
  loadB4(b0, wBase, ldw, 0);
  for (int k = 0; k < K; k += 64) {
    loadA(a1, aRow, k + 32);
    loadB4(b1, wBase, ldw, k + 32);
    __builtin_prefetch(aRow + k + 256, 0, 1);
    mma4(acc, a0, b0);
    if (k + 64 < K) {
      loadA(a0, aRow, k + 64);
      loadB4(b0, wBase, ldw, k + 64);
    }
    mma4(acc, a1, b1);
  }
  int rb = m0 + ((lane >> 4) << 3);
  #pragma unroll
  for (int j = 0; j < 4; ++j) {
    int col = n0 + j * 16 + lr;
    float bvv = bias ? bias[col] : 0.0f;
    #pragma unroll
    for (int g = 0; g < 8; ++g)
      C[(long long)(rb + g) * ldc + col] = acc[j][g] + bvv;
  }
}

// fused V-epilogue + attention prep: per (h, 64 tokens) block.
// emits v1h[h][n][d] (f16), and LDS-transposed v2t[h][d][n], xT[h][e][n] (f16)
__global__ __launch_bounds__(256) void attn_prep_kernel(
    const float* __restrict__ MU, const float* __restrict__ Qf, const float* __restrict__ Kf,
    const float* __restrict__ mask, const float* __restrict__ c_v, const float* __restrict__ c_attn,
    const float* __restrict__ sbv, const float* __restrict__ sbv2,
    _Float16* __restrict__ v1h, _Float16* __restrict__ v2t, _Float16* __restrict__ xT) {
  __shared__ float redA[64][4];
  __shared__ float redB[64][4];
  __shared__ float tV2[64][65];
  __shared__ float tX[64][65];
  int h = blockIdx.y;
  int n0 = blockIdx.x * 64;
  int t = threadIdx.x;
  int r = t >> 2;        // token row within tile
  int part = t & 3;      // 16-wide d chunk
  int n = n0 + r;
  float mu[16], sb[16];
  long long gbase = (long long)n * DIMC + h * DHC + part * 16;
  int sbase = h * DHC + part * 16;
  float sA = 0.0f, sB = 0.0f;
  #pragma unroll
  for (int i = 0; i < 16; ++i) {
    mu[i] = MU[gbase + i];
    sb[i] = sbv[sbase + i];
    sA += mu[i] * mu[i];
    sB += mu[i] * sb[i];
  }
  redA[r][part] = sA; redB[r][part] = sB;
  __syncthreads();
  float S_mm = redA[r][0] + redA[r][1] + redA[r][2] + redA[r][3];
  float S_ms = redB[r][0] + redB[r][1] + redB[r][2] + redB[r][3];
  float p, q, fp2, vv;
  stereo_post(S_mm, S_ms, c_v[h], sbv2[h], p, q, fp2, vv);
  float ka = c_attn[h];
  float pt = fmaxf(1.0f + ka * vv, EPSF);
  float gamma = fmaxf(2.0f / pt, EPSF);
  float g1 = gamma - 1.0f;
  float den = (g1 >= 0.0f ? 1.0f : -1.0f) * fmaxf(fabsf(g1), 1e-10f);
  float m = mask[n];
  float gdm = gamma / den * m;
  float ipt = 1.0f / pt;
  long long v1base = ((long long)h * NN + n) * DHC + part * 16;
  #pragma unroll
  for (int i = 0; i < 16; ++i) {
    float Vd = fp2 * (p * mu[i] + q * sb[i]);
    float v1 = eluf(Qf[gbase + i] * ipt) + 1.0f;
    float v2 = den * (eluf(Kf[gbase + i] * ipt) + 1.0f) * m;
    v1h[v1base + i] = (_Float16)v1;
    tV2[r][part * 16 + i] = v2;
    tX[r][part * 16 + i] = gdm * Vd;
  }
  __syncthreads();
  int d = r;  // now this thread owns channel d, writes 16 contiguous tokens
  long long obase = ((long long)(h * DHC + d)) * NN + n0 + part * 16;
  #pragma unroll
  for (int i = 0; i < 16; ++i) {
    v2t[obase + i] = (_Float16)tV2[part * 16 + i][d];
    xT[obase + i]  = (_Float16)tX[part * 16 + i][d];
  }
}

// per-head split-K GEMM: CTpart[slice][h][e][d] = sum_n xT[h][e][n]*v2t[h][d][n]
// (same double-buffered pipeline; Kslice is a multiple of 64)
__global__ __launch_bounds__(128) void gemm_ct_kernel(
    const _Float16* __restrict__ xT, const _Float16* __restrict__ v2t,
    float* __restrict__ CTpart, int Kslice) {
  int wave = threadIdx.x >> 5;
  int lane = threadIdx.x & 31;
  int slice = blockIdx.x, h = blockIdx.y;
  const _Float16* A = xT  + (long long)h * 64 * NN;
  const _Float16* W = v2t + (long long)h * 64 * NN;
  int m0 = wave * 16;
  int lr = lane & 15;
  int kh = (lane >> 4) << 3;
  long long k0 = (long long)slice * Kslice;
  const _Float16* aRow  = A + (long long)(m0 + lr) * NN + kh + k0;
  const _Float16* wBase = W + (long long)lr * NN + kh + k0;
  v8f z = {};
  v8f acc[4] = {z, z, z, z};
  AF a0, a1;
  AF b0[4], b1[4];
  loadA(a0, aRow, 0);
  loadB4(b0, wBase, NN, 0);
  for (int k = 0; k < Kslice; k += 64) {
    loadA(a1, aRow, k + 32);
    loadB4(b1, wBase, NN, k + 32);
    __builtin_prefetch(aRow + k + 256, 0, 1);
    mma4(acc, a0, b0);
    if (k + 64 < Kslice) {
      loadA(a0, aRow, k + 64);
      loadB4(b0, wBase, NN, k + 64);
    }
    mma4(acc, a1, b1);
  }
  float* C = CTpart + ((long long)slice * HC + h) * 4096;
  int rb = m0 + ((lane >> 4) << 3);
  #pragma unroll
  for (int j = 0; j < 4; ++j)
    #pragma unroll
    for (int g = 0; g < 8; ++g)
      C[(rb + g) * 64 + j * 16 + lr] = acc[j][g];
}

// deterministic split-K reduction -> CTh (f16), layout [h][e][d]
__global__ __launch_bounds__(256) void ct_reduce_kernel(
    const float* __restrict__ CTpart, _Float16* __restrict__ CTh) {
  int i = blockIdx.x * 256 + threadIdx.x;
  if (i >= HC * 4096) return;
  int h = i >> 12, rem = i & 4095;
  float s = 0.0f;
  for (int sl = 0; sl < NSLICE; ++sl)
    s += CTpart[((long long)sl * HC + h) * 4096 + rem];
  CTh[i] = (_Float16)s;
}

__global__ __launch_bounds__(256) void v2sum_kernel(
    const _Float16* __restrict__ v2t, float* __restrict__ v2sum) {
  int hd = blockIdx.x;         // h*64+d
  int t = threadIdx.x;
  const _Float16* p = v2t + (long long)hd * NN;
  float s = 0.0f;
  for (int n = t; n < NN; n += 256) s += (float)p[n];
  __shared__ float buf[256];
  buf[t] = s; __syncthreads();
  for (int off = 128; off > 0; off >>= 1) {
    if (t < off) buf[t] += buf[t + off];
    __syncthreads();
  }
  if (t == 0) v2sum[hd] = buf[0];
}

// Xo = D_inv*S; project; mobius_scalar_mul(0.5); project; then logmap0(c_ff) -> U2h f16
__global__ __launch_bounds__(64) void attn_epilogue_kernel(
    const float* __restrict__ S, const _Float16* __restrict__ v1h,
    const float* __restrict__ v2sum, const float* __restrict__ c_attn,
    const float* __restrict__ c_ff, _Float16* __restrict__ U2h) {
  int n = blockIdx.x, h = blockIdx.y, e = threadIdx.x;
  long long idx = (long long)n * DIMC + h * DHC + e;
  float s = S[idx];
  float v1 = (float)v1h[((long long)h * NN + n) * DHC + e];
  float vs = v2sum[h * DHC + e];
  float2 rr = blockReduce64x2(s * s, v1 * vs);
  float S2 = rr.x, Dn = rr.y;
  float Dinv = 1.0f / ((Dn == 0.0f) ? 1e-5f : Dn);
  float k = c_attn[h];
  float sk = sqrtf(fmaxf(fabsf(k), EPSF));
  float maxn = (1.0f - 1e-5f) / sk;
  float r0 = fabsf(Dinv) * sqrtf(fmaxf(S2, 0.0f));
  float n1 = fmaxf(r0, EPSF);
  float f1 = (n1 > maxn) ? maxn / n1 : 1.0f;
  float xn = fmaxf(f1 * r0, EPSF);
  float art = artk_sk(xn, sk);
  float tt = tanhf(0.5f * art * sk) / sk;
  float sc2 = tt / xn;
  float r2 = fabsf(tt);
  float n3 = fmaxf(r2, EPSF);
  float f3 = (n3 > maxn) ? maxn / n3 : 1.0f;
  float rA = f3 * r2;
  float kff = c_ff[h];
  float skf = sqrtf(fmaxf(fabsf(kff), EPSF));
  float yn = fmaxf(rA, EPSF);
  float artf = artk_sk(yn, skf) / yn;
  float total = Dinv * f1 * sc2 * f3 * artf;
  U2h[idx] = (_Float16)(total * s);
}

// final stereo_linear epilogue -> f32 output
__global__ __launch_bounds__(64) void final_kernel(
    const float* __restrict__ MU2, const float* __restrict__ sbff,
    const float* __restrict__ sb2ff, const float* __restrict__ c_ff,
    float* __restrict__ out) {
  int n = blockIdx.x, h = blockIdx.y, d = threadIdx.x;
  long long idx = (long long)n * DIMC + h * DHC + d;
  float mu = MU2[idx];
  float sb = sbff[h * DHC + d];
  float2 rr = blockReduce64x2(mu * mu, mu * sb);
  float p, q, fp2, vv;
  stereo_post(rr.x, rr.y, c_ff[h], sb2ff[h], p, q, fp2, vv);
  out[idx] = fp2 * (p * mu + q * sb);
}

extern "C" void kernel_launch(void* const* d_in, const int* in_sizes, int n_in,
                              void* d_out, int out_size, void* d_ws, size_t ws_size,
                              hipStream_t stream) {
  (void)in_sizes; (void)n_in; (void)out_size; (void)ws_size;
  const float* X     = (const float*)d_in[0];
  const float* mask  = (const float*)d_in[1];
  const float* Wq    = (const float*)d_in[2];
  const float* bq    = (const float*)d_in[3];
  const float* Wk    = (const float*)d_in[4];
  const float* bk    = (const float*)d_in[5];
  const float* Wv    = (const float*)d_in[6];
  const float* bvp   = (const float*)d_in[7];
  const float* c_v   = (const float*)d_in[8];
  const float* c_at  = (const float*)d_in[9];
  const float* Wff   = (const float*)d_in[10];
  const float* bff   = (const float*)d_in[11];
  const float* c_ff  = (const float*)d_in[12];
  float* out = (float*)d_out;

  char* ws = (char*)d_ws;
  const long long MB = 1024ll * 1024;
  // aliasing is safe: consumers of the first name finish (stream order)
  // before producers of the second name run.
  _Float16* Xh   = (_Float16*)(ws + 0);        // 32MB  -> reused as xT
  _Float16* xT   = Xh;
  _Float16* Uh   = (_Float16*)(ws + 32 * MB);  // 32MB  -> reused as v1h
  _Float16* v1h  = Uh;
  float*    Qf   = (float*)(ws + 64 * MB);     // 64MB  -> reused as U2h
  _Float16* U2h  = (_Float16*)Qf;
  float*    Kf   = (float*)(ws + 128 * MB);    // 64MB  -> reused as MU2
  float*    MU2  = Kf;
  float*    MU   = (float*)(ws + 192 * MB);    // 64MB  -> reused as S
  float*    S    = MU;
  _Float16* v2t  = (_Float16*)(ws + 256 * MB); // 32MB
  _Float16* Wqh  = (_Float16*)(ws + 288 * MB); // 2MB each
  _Float16* Wkh  = (_Float16*)(ws + 290 * MB);
  _Float16* Wvh  = (_Float16*)(ws + 292 * MB);
  _Float16* Wffh = (_Float16*)(ws + 294 * MB);
  float*    sbv   = (float*)(ws + 296 * MB);
  float*    sbv2  = (float*)(ws + 296 * MB + 4096);
  float*    sbf   = (float*)(ws + 296 * MB + 8192);
  float*    sbf2  = (float*)(ws + 296 * MB + 12288);
  float*    v2sum = (float*)(ws + 296 * MB + 16384);
  _Float16* CTh   = (_Float16*)(ws + 296 * MB + 32768);  // 128KB
  float*    CTpart= (float*)(ws + 297 * MB);             // 8MB

  const int WE = DIMC * DIMC;
  cvt_f32_f16<<<(WE + 255) / 256, 256, 0, stream>>>(Wq, Wqh, WE);
  cvt_f32_f16<<<(WE + 255) / 256, 256, 0, stream>>>(Wk, Wkh, WE);
  cvt_f32_f16<<<(WE + 255) / 256, 256, 0, stream>>>(Wv, Wvh, WE);
  cvt_f32_f16<<<(WE + 255) / 256, 256, 0, stream>>>(Wff, Wffh, WE);

  sb_kernel<<<dim3(HC, 2), 64, 0, stream>>>(bvp, c_v, bff, c_ff, sbv, sbv2, sbf, sbf2);

  logmap_x_kernel<<<dim3(NN, HC), 64, 0, stream>>>(X, c_v, Xh, Uh);

  // Q, K, MU GEMMs (16384x1024x1024, f16 WMMA, f32 accum)
  gemm_wmma_kernel<<<dim3(NN / 64, DIMC / 64, 1), 128, 0, stream>>>(
      Xh, DIMC, 0, Wqh, DIMC, 0, Qf, DIMC, 0, bq, NN, DIMC);
  gemm_wmma_kernel<<<dim3(NN / 64, DIMC / 64, 1), 128, 0, stream>>>(
      Xh, DIMC, 0, Wkh, DIMC, 0, Kf, DIMC, 0, bk, NN, DIMC);
  gemm_wmma_kernel<<<dim3(NN / 64, DIMC / 64, 1), 128, 0, stream>>>(
      Uh, DIMC, 0, Wvh, DIMC, 0, MU, DIMC, 0, nullptr, NN, DIMC);

  attn_prep_kernel<<<dim3(NN / 64, HC), 256, 0, stream>>>(
      MU, Qf, Kf, mask, c_v, c_at, sbv, sbv2, v1h, v2t, xT);

  // ctx^T per head with deterministic split-K (32 slices of 512)
  gemm_ct_kernel<<<dim3(NSLICE, HC), 128, 0, stream>>>(xT, v2t, CTpart, NN / NSLICE);
  ct_reduce_kernel<<<(HC * 4096 + 255) / 256, 256, 0, stream>>>(CTpart, CTh);

  v2sum_kernel<<<HC * DHC, 256, 0, stream>>>(v2t, v2sum);

  // S[n][h*64+e] = sum_d v1[h,n,d]*ctx[h,d,e]  (batched over heads)
  gemm_wmma_kernel<<<dim3(NN / 64, 1, HC), 128, 0, stream>>>(
      v1h, DHC, 64ll * NN, CTh, DHC, 4096, S, DIMC, 64, nullptr, NN, DHC);

  attn_epilogue_kernel<<<dim3(NN, HC), 64, 0, stream>>>(S, v1h, v2sum, c_at, c_ff, U2h);

  gemm_wmma_kernel<<<dim3(NN / 64, DIMC / 64, 1), 128, 0, stream>>>(
      U2h, DIMC, 0, Wffh, DIMC, 0, MU2, DIMC, 0, nullptr, NN, DIMC);

  final_kernel<<<dim3(NN, HC), 64, 0, stream>>>(MU2, sbf, sbf2, c_ff, out);
}
